// Flash_attn_16131897164211
// MI455X (gfx1250) — matrix operand order repaired and checked by reading, not compiled, not run on hardware
//
#include <hip/hip_runtime.h>
#include <hip/hip_bf16.h>

// ---------------------------------------------------------------------------
// MI455X (gfx1250) implementation of the windowed quadratic+linear attention.
// All GEMMs use v_wmma_f32_16x16x32_bf16 (wave32, 16 waves / 512-thread block).
// bf16 intermediates halve HBM traffic (workload is memory-bound: ~100 GFLOP
// vs ~0.7 GB traffic at 23.3 TB/s); f32 accumulation throughout.
// Tile staging into LDS uses GLOBAL_LOAD_ASYNC_TO_LDS_B128 (ASYNCcnt path).
// ---------------------------------------------------------------------------

typedef __bf16 bf16;
typedef __attribute__((ext_vector_type(16))) __bf16 v16bf;
typedef __attribute__((ext_vector_type(8)))  float  v8f;

__device__ __forceinline__ v8f wmma_bf16(v16bf a, v16bf b, v8f c) {
  // D = A(16x32) * B(32x16) + C(16x16), f32 accumulate
  return __builtin_amdgcn_wmma_f32_16x16x32_bf16(
      false, a, false, b, (short)0, c, false, false);
}

// fast silu: x * sigmoid(x) with v_rcp_f32 instead of IEEE divide
__device__ __forceinline__ float fast_silu(float x) {
  return x * __builtin_amdgcn_rcpf(1.f + __expf(-x));
}

// async 16-byte copy global -> LDS (ISA 15.18.3 op 98, tracked on ASYNCcnt).
// lds_dst: generic pointer to __shared__; low 32 bits = wave-relative LDS addr.
__device__ __forceinline__ void async_cp_b128(void* lds_dst, const void* gsrc) {
  unsigned ldsa = (unsigned)(unsigned long long)lds_dst;
  asm volatile("global_load_async_to_lds_b128 %0, %1, off"
               :: "v"(ldsa), "v"(gsrc) : "memory");
}
__device__ __forceinline__ void async_wait0() {
  asm volatile("s_wait_asynccnt 0" ::: "memory");
}

// A fragment: row-major source, lane (ln,hi) holds row m; a[j]=A[m][k0+hi*16+j]
// (contiguous k order, matching the B fragments)
__device__ __forceinline__ v16bf load_a_row(const bf16* A, int lda, int m, int k0, int hi) {
  v16bf a;
  const bf16* p = A + (size_t)m * lda + k0 + hi * 16;
#pragma unroll
  for (int j = 0; j < 8; ++j) { a[j] = p[j]; a[8 + j] = p[8 + j]; }
  return a;
}

// B fragment: row-major K x N source; lane (ln,hi) holds col n; b[j]=B[k0+hi*16+j][n]
__device__ __forceinline__ v16bf load_b_row(const bf16* B, int ldb, int k0, int n, int hi) {
  v16bf b;
  const bf16* p = B + (size_t)(k0 + hi * 16) * ldb + n;
#pragma unroll
  for (int j = 0; j < 16; ++j) { b[j] = p[(size_t)j * ldb]; }
  return b;
}

// A fragment from a transposed source: A[m][k] = S[k][m] (S row-major, ld=lds)
__device__ __forceinline__ v16bf load_a_trans(const bf16* S, int lds, int m, int k0, int hi) {
  v16bf a;
#pragma unroll
  for (int j = 0; j < 8; ++j) {
    a[j]     = S[(size_t)(k0 + hi * 16 + j) * lds + m];
    a[8 + j] = S[(size_t)(k0 + hi * 16 + 8 + j) * lds + m];
  }
  return a;
}

// ---------------------------------------------------------------------------
// Geometry constants
// ---------------------------------------------------------------------------
#define BIMG 64
#define HH   32
#define WW   64
#define CC   192
#define EE   384      // e = 2C
#define SS   128
#define NTOK 64       // tokens per window (8x8)
#define NWIN 2048     // 64 * (32/8) * (64/8)
#define TTOT 131072   // NWIN * NTOK
#define UVB  896      // 2e + S

// ---------------------------------------------------------------------------
// Kernel 0: cast weights f32->bf16, zero KV accumulator
// ---------------------------------------------------------------------------
__global__ void k_prep(const float* Wuv, const float* Wp,
                       bf16* Wuv_bf, bf16* Wp_bf, float* KV) {
  int i = blockIdx.x * blockDim.x + threadIdx.x;
  if (i < CC * UVB) Wuv_bf[i] = (bf16)Wuv[i];
  if (i < EE * CC)  Wp_bf[i]  = (bf16)Wp[i];
  if (i < SS * EE)  KV[i]     = 0.f;
}

// ---------------------------------------------------------------------------
// Kernel 1: uvb = silu(window_partition(x) @ W_uv + b_uv); split into u,v,base
// grid = (2048 windows, 14 column tiles of 64); block = 256 (8 waves)
// ---------------------------------------------------------------------------
__global__ __launch_bounds__(256) void k_gemm1(
    const float* __restrict__ x, const bf16* __restrict__ Wuv_bf,
    const float* __restrict__ b_uv,
    bf16* __restrict__ u_bf, bf16* __restrict__ v_bf, float* __restrict__ base_f) {
  __shared__ bf16 Al[64 * 200];                 // 64 tokens x 192 ch (padded)
  const int win = blockIdx.x, ct = blockIdx.y, tid = threadIdx.x;
  const int hwi = (win >> 3) & 3, wwi = win & 7, bimg = win >> 5;

  for (int idx = tid; idx < NTOK * CC; idx += 256) {
    int t = idx / CC, c = idx - t * CC;
    int gh = hwi * 8 + (t >> 3), gw = wwi * 8 + (t & 7);
    Al[t * 200 + c] = (bf16)x[(((size_t)bimg * HH + gh) * WW + gw) * CC + c];
  }
  __syncthreads();

  const int wave = tid >> 5, lane = tid & 31, hi = lane >> 4, ln = lane & 15;
  const int mt = wave >> 1, ntb = (wave & 1) * 2;  // wave -> 16x16 tiles (mt, ntb..ntb+1)
  v8f acc[2] = {};
  const int m = mt * 16 + ln;
  const int col0 = ct * 64 + (ntb + 0) * 16 + ln;
  const int col1 = ct * 64 + (ntb + 1) * 16 + ln;

  for (int kb = 0; kb < CC; kb += 32) {
    v16bf a = load_a_row(Al, 200, m, kb, hi);
    __builtin_prefetch(Wuv_bf + (size_t)(kb + 32) * UVB + col0, 0, 1);
    v16bf b0 = load_b_row(Wuv_bf, UVB, kb, col0, hi);
    v16bf b1 = load_b_row(Wuv_bf, UVB, kb, col1, hi);
    acc[0] = wmma_bf16(a, b0, acc[0]);
    acc[1] = wmma_bf16(a, b1, acc[1]);
  }

#pragma unroll
  for (int pt = 0; pt < 2; ++pt) {
    const int gcol = ct * 64 + (ntb + pt) * 16 + ln;
    const float bias = b_uv[gcol];
#pragma unroll
    for (int r = 0; r < 8; ++r) {
      const int row = mt * 16 + hi * 8 + r;
      const size_t tok = (size_t)win * NTOK + row;
      float vv = fast_silu(acc[pt][r] + bias);
      if (gcol < EE)            u_bf[tok * EE + gcol]            = (bf16)vv;
      else if (gcol < 2 * EE)   v_bf[tok * EE + (gcol - EE)]     = (bf16)vv;
      else                      base_f[tok * SS + (gcol - 2*EE)] = vv;
    }
  }
}

// ---------------------------------------------------------------------------
// Kernel 2: gates + 2D RoPE -> qq, qk (window coords), lq, lk (global coords)
// one thread per (token, channel-pair)
// ---------------------------------------------------------------------------
__device__ __forceinline__ void rope_store(bf16* dst, size_t tok, int c0, int c1,
                                           float xe, float xo, float pos, float inv) {
  const float ang = pos * inv;
  const float cs = __cosf(ang), sn = __sinf(ang);
  dst[tok * SS + c0] = (bf16)(xe * cs - xo * sn);
  dst[tok * SS + c1] = (bf16)(xe * sn + xo * cs);
}

__global__ void k_rope(const float* __restrict__ base_f,
                       const float* gqg, const float* gqb,
                       const float* gkg, const float* gkb,
                       const float* lqg, const float* lqb,
                       const float* lkg, const float* lkb,
                       bf16* qq, bf16* qk, bf16* lq, bf16* lk) {
  const size_t gid = (size_t)blockIdx.x * blockDim.x + threadIdx.x;
  if (gid >= (size_t)TTOT * 64) return;
  const int p = (int)(gid & 63);
  const size_t tok = gid >> 6;
  const int win = (int)(tok >> 6), idx = (int)(tok & 63);
  const int i = idx >> 3, j = idx & 7;
  const int hwi = (win >> 3) & 3, wwi = win & 7;
  const int gh = hwi * 8 + i, gw = wwi * 8 + j;

  const int half = p >> 5, f = p & 31;
  const int c0 = half * 64 + 2 * f, c1 = c0 + 1;
  // inv_freq = 10000^(-2f/64)   (d = 64 per half)
  const float inv = __expf(-(float)(2 * f) * (9.210340371976184f / 64.f));
  const float posL = half ? (float)j : (float)i;
  const float posG = half ? (float)gw : (float)gh;

  const float b0 = base_f[tok * SS + c0], b1 = base_f[tok * SS + c1];
  rope_store(qq, tok, c0, c1, b0 * gqg[c0] + gqb[c0], b1 * gqg[c1] + gqb[c1], posL, inv);
  rope_store(qk, tok, c0, c1, b0 * gkg[c0] + gkb[c0], b1 * gkg[c1] + gkb[c1], posL, inv);
  rope_store(lq, tok, c0, c1, b0 * lqg[c0] + lqb[c0], b1 * lqg[c1] + lqb[c1], posG, inv);
  rope_store(lk, tok, c0, c1, b0 * lkg[c0] + lkb[c0], b1 * lkg[c1] + lkb[c1], posG, inv);
}

// ---------------------------------------------------------------------------
// Kernel 3: per-window attention.
//   attn = relu((qq/64)@qk^T + rel_bias)^2 ; quad = attn @ v
// one block (16 waves) per window; qq/qk staged via async LDS DMA
// ---------------------------------------------------------------------------
__global__ __launch_bounds__(512) void k_attn(
    const bf16* __restrict__ qq, const bf16* __restrict__ qk,
    const bf16* __restrict__ v_bf, const float* __restrict__ rel_bias,
    bf16* __restrict__ quad_bf) {
  __shared__ bf16 qql[64 * 136];
  __shared__ bf16 qkl[64 * 136];
  __shared__ bf16 at[64 * 72];
  __shared__ float rb[225];
  const int win = blockIdx.x, tid = threadIdx.x;
  const int wave = tid >> 5, lane = tid & 31, hi = lane >> 4, ln = lane & 15;

  // 64 rows x 256B payload = 1024 b128 chunks per matrix; rows padded to 272B
  const bf16* qsrc = qq + (size_t)win * NTOK * SS;
  const bf16* ksrc = qk + (size_t)win * NTOK * SS;
  for (int idx = tid; idx < NTOK * 16; idx += 512) {
    int r = idx >> 4, c = idx & 15;               // chunk c = 8 bf16 = 16B
    async_cp_b128(&qql[r * 136 + c * 8], qsrc + r * SS + c * 8);
    async_cp_b128(&qkl[r * 136 + c * 8], ksrc + r * SS + c * 8);
  }
  if (tid < 225) rb[tid] = rel_bias[tid];
  async_wait0();
  __syncthreads();

  // ---- phase 1: 64x64 attention scores, 16 waves <-> 16 tiles
  {
    const int mt = wave >> 2, nt = wave & 3;
    const int m = mt * 16 + ln, n = nt * 16 + ln;
    v8f acc = {};
#pragma unroll
    for (int kb = 0; kb < SS; kb += 32) {
      v16bf a = load_a_row(qql, 136, m, kb, hi);
      v16bf b;  // B[k][n] = qk[n][k] : contiguous per lane
      const bf16* bp = &qkl[n * 136 + kb + hi * 16];
#pragma unroll
      for (int j = 0; j < 16; ++j) b[j] = bp[j];
      acc = wmma_bf16(a, b, acc);
    }
    const float scale = 1.f / 64.f;
#pragma unroll
    for (int r = 0; r < 8; ++r) {
      const int row = mt * 16 + hi * 8 + r, coln = nt * 16 + ln;
      const int i1 = row >> 3, j1 = row & 7, i2 = coln >> 3, j2 = coln & 7;
      float vv = acc[r] * scale + rb[(i1 - i2 + 7) * 15 + (j1 - j2 + 7)];
      vv = fmaxf(vv, 0.f); vv *= vv;
      at[row * 72 + coln] = (bf16)vv;
    }
  }
  __syncthreads();

  // ---- phase 2: quad = attn(64x64) @ v(64x384); 96 tiles / 16 waves
  {
    const int mt = wave & 3, ntb = wave >> 2;
    const int m = mt * 16 + ln;
#pragma unroll
    for (int t = 0; t < 6; ++t) {
      const int nt = ntb + 4 * t;
      const int col = nt * 16 + ln;
      v8f acc = {};
#pragma unroll
      for (int kb = 0; kb < NTOK; kb += 32) {
        v16bf a = load_a_row(at, 72, m, kb, hi);
        v16bf b = load_b_row(v_bf + (size_t)win * NTOK * EE, EE, kb, col, hi);
        acc = wmma_bf16(a, b, acc);
      }
#pragma unroll
      for (int r = 0; r < 8; ++r) {
        const size_t tok = (size_t)win * NTOK + mt * 16 + hi * 8 + r;
        quad_bf[tok * EE + col] = (bf16)acc[r];
      }
    }
  }
}

// ---------------------------------------------------------------------------
// Kernel 4: KV[s][e] = sum_t lk[t][s] * v[t][e] / 2048   (global reduction)
// 256 blocks x 16 waves; each block accumulates 512 tokens, then atomicAdd
// ---------------------------------------------------------------------------
__global__ __launch_bounds__(512) void k_kv(
    const bf16* __restrict__ lk, const bf16* __restrict__ v_bf,
    float* __restrict__ KV) {
  __shared__ bf16 lkl[32 * 132];
  const int tid = threadIdx.x;
  const int wave = tid >> 5, lane = tid & 31, hi = lane >> 4, ln = lane & 15;
  const int mt = wave >> 1;                     // s-tile 0..7
  v8f acc[12] = {};

  const int chunk0 = blockIdx.x * 16;
  for (int ch = 0; ch < 16; ++ch) {
    const size_t tok0 = (size_t)(chunk0 + ch) * 32;
    for (int idx = tid; idx < 32 * SS; idx += 512) {
      int r = idx >> 7, c = idx & 127;
      lkl[r * 132 + c] = lk[(tok0 + r) * SS + c];
    }
    __syncthreads();
    const int m = mt * 16 + ln;
    v16bf a = load_a_trans(lkl, 132, m, 0, hi);  // A = lk^T chunk (K=32)
#pragma unroll
    for (int t = 0; t < 12; ++t) {
      const int col = ((wave & 1) + 2 * t) * 16 + ln;
      v16bf b = load_b_row(v_bf + tok0 * EE, EE, 0, col, hi);
      acc[t] = wmma_bf16(a, b, acc[t]);
    }
    __syncthreads();
  }
  const float sc = 1.f / 2048.f;                 // 1 / (N * nW)
#pragma unroll
  for (int t = 0; t < 12; ++t) {
    const int col = ((wave & 1) + 2 * t) * 16 + ln;
#pragma unroll
    for (int r = 0; r < 8; ++r) {
      const int row = mt * 16 + hi * 8 + r;
      atomicAdd(&KV[row * EE + col], acc[t][r] * sc);
    }
  }
}

__global__ void k_kvcast(const float* KV, bf16* KVb) {
  int i = blockIdx.x * blockDim.x + threadIdx.x;
  if (i < SS * EE) KVb[i] = (bf16)KV[i];
}

// ---------------------------------------------------------------------------
// Kernel 5a: A2 = u * (quad + lq @ KV)          (pre-projection activations)
// ---------------------------------------------------------------------------
__global__ __launch_bounds__(512) void k_lin(
    const bf16* __restrict__ lq, const bf16* __restrict__ KVb,
    const bf16* __restrict__ quad_bf, const bf16* __restrict__ u_bf,
    bf16* __restrict__ A2) {
  __shared__ bf16 lql[64 * 136];
  const int win = blockIdx.x, tid = threadIdx.x;
  const int wave = tid >> 5, lane = tid & 31, hi = lane >> 4, ln = lane & 15;

  const bf16* lsrc = lq + (size_t)win * NTOK * SS;
  for (int idx = tid; idx < NTOK * 16; idx += 512) {
    int r = idx >> 4, c = idx & 15;
    async_cp_b128(&lql[r * 136 + c * 8], lsrc + r * SS + c * 8);
  }
  async_wait0();
  __syncthreads();

  const int mt = wave & 3, ntb = wave >> 2;
  const int m = mt * 16 + ln;
#pragma unroll
  for (int t = 0; t < 6; ++t) {
    const int col = (ntb + 4 * t) * 16 + ln;
    v8f acc = {};
#pragma unroll
    for (int kb = 0; kb < SS; kb += 32) {
      v16bf a = load_a_row(lql, 136, m, kb, hi);
      v16bf b = load_b_row(KVb, EE, kb, col, hi);   // KV is L2-resident (96 KB)
      acc = wmma_bf16(a, b, acc);
    }
#pragma unroll
    for (int r = 0; r < 8; ++r) {
      const size_t tok = (size_t)win * NTOK + mt * 16 + hi * 8 + r;
      float vv = acc[r] + (float)quad_bf[tok * EE + col];
      vv *= (float)u_bf[tok * EE + col];
      A2[tok * EE + col] = (bf16)vv;
    }
  }
}

// ---------------------------------------------------------------------------
// Kernel 5b: out = window_reverse(A2 @ W_proj + b_proj)
// ---------------------------------------------------------------------------
__global__ __launch_bounds__(512) void k_proj(
    const bf16* __restrict__ A2, const bf16* __restrict__ Wp_bf,
    const float* __restrict__ b_proj, float* __restrict__ out) {
  __shared__ bf16 al[64 * 392];
  const int win = blockIdx.x, tid = threadIdx.x;
  const int wave = tid >> 5, lane = tid & 31, hi = lane >> 4, ln = lane & 15;
  const int hwi = (win >> 3) & 3, wwi = win & 7, bimg = win >> 5;

  // 64 rows x 768B payload = 48 b128 chunks per row; rows padded to 784B
  const bf16* asrc = A2 + (size_t)win * NTOK * EE;
  for (int idx = tid; idx < NTOK * 48; idx += 512) {
    int r = idx / 48, c = idx - r * 48;
    async_cp_b128(&al[r * 392 + c * 8], asrc + r * EE + c * 8);
  }
  async_wait0();
  __syncthreads();

  const int mt = wave & 3, ntb = wave >> 2;
  const int m = mt * 16 + ln;
#pragma unroll
  for (int t = 0; t < 3; ++t) {
    const int nt = ntb * 3 + t;                    // 0..11 (N=192)
    const int col = nt * 16 + ln;
    v8f acc = {};
    for (int kb = 0; kb < EE; kb += 32) {
      v16bf a = load_a_row(al, 392, m, kb, hi);
      __builtin_prefetch(Wp_bf + (size_t)(kb + 32) * CC + col, 0, 1);
      v16bf b = load_b_row(Wp_bf, CC, kb, col, hi);
      acc = wmma_bf16(a, b, acc);
    }
    const float bias = b_proj[col];
#pragma unroll
    for (int r = 0; r < 8; ++r) {
      const int row = mt * 16 + hi * 8 + r;
      const int gh = hwi * 8 + (row >> 3), gw = wwi * 8 + (row & 7);
      out[(((size_t)bimg * HH + gh) * WW + gw) * CC + col] = acc[r] + bias;
    }
  }
}

// ---------------------------------------------------------------------------
// Launcher
// ---------------------------------------------------------------------------
extern "C" void kernel_launch(void* const* d_in, const int* in_sizes, int n_in,
                              void* d_out, int out_size, void* d_ws, size_t ws_size,
                              hipStream_t stream) {
  (void)in_sizes; (void)n_in; (void)out_size; (void)ws_size;
  const float* x    = (const float*)d_in[0];
  const float* Wuv  = (const float*)d_in[1];
  const float* buv  = (const float*)d_in[2];
  const float* gqg  = (const float*)d_in[3];
  const float* gqb  = (const float*)d_in[4];
  const float* gkg  = (const float*)d_in[5];
  const float* gkb  = (const float*)d_in[6];
  const float* lqg  = (const float*)d_in[7];
  const float* lqb  = (const float*)d_in[8];
  const float* lkg  = (const float*)d_in[9];
  const float* lkb  = (const float*)d_in[10];
  const float* relb = (const float*)d_in[11];
  const float* Wp   = (const float*)d_in[12];
  const float* bp   = (const float*)d_in[13];
  float* out = (float*)d_out;

  char* p = (char*)d_ws;
  auto alloc = [&](size_t bytes) -> void* {
    void* r = (void*)p; p += (bytes + 255) & ~(size_t)255; return r;
  };
  bf16*  Wuv_bf = (bf16*) alloc((size_t)CC * UVB * 2);
  bf16*  Wp_bf  = (bf16*) alloc((size_t)EE * CC * 2);
  float* KV     = (float*)alloc((size_t)SS * EE * 4);
  bf16*  KVb    = (bf16*) alloc((size_t)SS * EE * 2);
  bf16*  u_bf   = (bf16*) alloc((size_t)TTOT * EE * 2);
  bf16*  v_bf   = (bf16*) alloc((size_t)TTOT * EE * 2);
  float* base_f = (float*)alloc((size_t)TTOT * SS * 4);
  bf16*  qq     = (bf16*) alloc((size_t)TTOT * SS * 2);   // \  contiguous: the
  bf16*  qk     = (bf16*) alloc((size_t)TTOT * SS * 2);   //  } qq|qk|lk region
  bf16*  lk     = (bf16*) alloc((size_t)TTOT * SS * 2);   // /  is reused as A2
  bf16*  lq     = (bf16*) alloc((size_t)TTOT * SS * 2);
  bf16*  quadb  = (bf16*) alloc((size_t)TTOT * EE * 2);
  bf16*  A2     = qq;  // 3 * TTOT*SS*2 == TTOT*EE*2 ; qq/qk/lk dead by k_lin

  k_prep<<<(CC * UVB + 255) / 256, 256, 0, stream>>>(Wuv, Wp, Wuv_bf, Wp_bf, KV);
  k_gemm1<<<dim3(NWIN, UVB / 64), 256, 0, stream>>>(x, Wuv_bf, buv, u_bf, v_bf, base_f);
  k_rope<<<(TTOT * 64) / 256, 256, 0, stream>>>(base_f, gqg, gqb, gkg, gkb,
                                                lqg, lqb, lkg, lkb, qq, qk, lq, lk);
  k_attn<<<NWIN, 512, 0, stream>>>(qq, qk, v_bf, relb, quadb);
  k_kv<<<256, 512, 0, stream>>>(lk, v_bf, KV);
  k_kvcast<<<(SS * EE + 255) / 256, 256, 0, stream>>>(KV, KVb);
  k_lin<<<NWIN, 512, 0, stream>>>(lq, KVb, quadb, u_bf, A2);
  k_proj<<<NWIN, 512, 0, stream>>>(A2, Wp_bf, bp, out);
}
